// SenticGCN_BERT_30331059044475
// MI455X (gfx1250) — compile-verified, matrix operand order checked by reading, not run on hardware
//
#include <hip/hip_runtime.h>
#include <hip/hip_bf16.h>

// ---------------------------------------------------------------------------
// SenticGCN-BERT forward, CDNA5/gfx1250.
// B=128, S=256, D=768, P=3.
// GEMMs on v_wmma_f32_16x16x32_bf16 (wave32), 128x32 output tile per block,
// 2 WMMAs per wave per K-step (shared A fragment). Intermediate H kept bf16,
// transposed [B, D, S] so GEMM2's B-panel is K-contiguous and is DMA'd into
// LDS with global_load_async_to_lds_b64 (ASYNCcnt path).
// ---------------------------------------------------------------------------

constexpr int NB = 128;   // batch
constexpr int S  = 256;   // seq len
constexpr int D  = 768;   // hidden
constexpr int P  = 3;     // polarities

constexpr int TM  = 128;  // M rows per workgroup tile
constexpr int TN  = 32;   // N cols per workgroup tile (2 x 16 per wave)
constexpr int TK  = 32;   // K step (matches 16x16x32 bf16 WMMA)
constexpr int LDA = TK + 2; // A pad: row stride 68B = 17 banks
constexpr int LDB = TK + 4; // B pad: row stride 72B -> 8B-aligned async writes

typedef __attribute__((ext_vector_type(16))) __bf16 v16bf;
typedef __attribute__((ext_vector_type(8)))  float  v8f;

__device__ __forceinline__ __bf16 cvt_bf16(float f) {
  unsigned u = __builtin_bit_cast(unsigned, f);
  u += 0x7FFFu + ((u >> 16) & 1u);           // round-to-nearest-even
  unsigned short h = (unsigned short)(u >> 16);
  return __builtin_bit_cast(__bf16, h);
}

// low 32 bits of a flat LDS-aperture address == addrspace(3) byte offset
__device__ __forceinline__ unsigned lds_off32(const void* p) {
  return (unsigned)(size_t)p;
}

// ---------------------------------------------------------------------------
// Kernel 1: per-batch lengths -> position weights pw[b,s], start[b], end[b].
// ---------------------------------------------------------------------------
__global__ __launch_bounds__(S) void meta_kernel(
    const int* __restrict__ ti, const int* __restrict__ ai,
    const int* __restrict__ li, float* __restrict__ pw,
    int* __restrict__ stA, int* __restrict__ enA) {
  const int b = blockIdx.x, j = threadIdx.x;
  __shared__ int cnt[3];
  if (j < 3) cnt[j] = 0;
  __syncthreads();
  if (ti[b * S + j] != 0) atomicAdd(&cnt[0], 1);
  if (ai[b * S + j] != 0) atomicAdd(&cnt[1], 1);
  if (li[b * S + j] != 0) atomicAdd(&cnt[2], 1);
  __syncthreads();
  const int tl = cnt[0], al = cnt[1], ll = cnt[2];
  const int st = ll, en = ll + al - 1;
  const float ctx = (float)(tl - al);
  const float jf = (float)j;
  float w;
  if (j < st)       w = 1.0f - ((float)st - jf) / ctx;
  else if (j <= en) w = 0.0f;
  else if (j < tl)  w = 1.0f - (jf - (float)en) / ctx;
  else              w = 0.0f;
  pw[b * S + j] = w;
  if (j == 0) { stA[b] = st; enA[b] = en; }
}

// ---------------------------------------------------------------------------
// Kernel 2: denom[b,s] = sum_t adj[b,s,t] + 1.  One wave per row.
// ---------------------------------------------------------------------------
__global__ __launch_bounds__(256) void denom_kernel(
    const float* __restrict__ adj, float* __restrict__ denomv) {
  const int lane = threadIdx.x & 31, wave = threadIdx.x >> 5;
  const int row = blockIdx.x * 8 + wave;            // [0, NB*S)
  const float* p = adj + (size_t)row * S;
  float s = 0.0f;
  #pragma unroll
  for (int i = 0; i < S; i += 32) s += p[i + lane];
  #pragma unroll
  for (int off = 16; off; off >>= 1) s += __shfl_xor(s, off, 32);
  if (lane == 0) denomv[row] = s + 1.0f;
}

// ---------------------------------------------------------------------------
// Wave32 16-bit WMMA fragment layouts (CDNA5):
// A 16x32: lane<16 -> M=lane, K in {0..7,16..23}; lane>=16 -> K in {8..15,24..31}.
// B 32x16 mirrored (N = lane&15, same K split). C/D: VGPR r -> M = r (+8 hi lanes).
// ---------------------------------------------------------------------------

// GEMM1: H[b][e][t] (bf16, transposed) = pw(m) * X[m,:] @ W[:, e], m = b*S + t.
__global__ __launch_bounds__(256) void gcn_xw_kernel(
    const float* __restrict__ X, const float* __restrict__ W,
    const float* __restrict__ pw, __bf16* __restrict__ Ht) {
  __shared__ __bf16 As[TM][LDA];
  __shared__ __bf16 Bs[TN][LDB];
  const int tid  = threadIdx.x;
  const int lane = tid & 31, wave = tid >> 5;
  const int Mbase = blockIdx.x * TM;
  const int n0    = blockIdx.y * TN;

  // A staging: 2 threads per row, 16 floats each (4x float4)
  const int arow  = tid >> 1;
  const int acol0 = (tid & 1) * 16;
  const float pwv = pw[Mbase + arow];
  const float* Xrow = X + (size_t)(Mbase + arow) * D;

  // B staging: Bs[n][k] = W[(k0+k)*D + n0+n], 32x32 tile, 4 elems/thread
  const int bn = tid & 31;
  const int bk = tid >> 5;   // 0..7

  const int kb   = (lane < 16) ? 0 : 8;
  const int mloc = wave * 16 + (lane & 15);
  const int nloc = lane & 15;

  v8f c0 = {}, c1 = {};
  for (int k0 = 0; k0 < D; k0 += TK) {
    if (k0 + TK < D) __builtin_prefetch(Xrow + k0 + TK + acol0, 0, 0);
    #pragma unroll
    for (int q = 0; q < 4; ++q) {
      float4 v = *(const float4*)(Xrow + k0 + acol0 + 4 * q);
      As[arow][acol0 + 4 * q + 0] = cvt_bf16(v.x * pwv);
      As[arow][acol0 + 4 * q + 1] = cvt_bf16(v.y * pwv);
      As[arow][acol0 + 4 * q + 2] = cvt_bf16(v.z * pwv);
      As[arow][acol0 + 4 * q + 3] = cvt_bf16(v.w * pwv);
    }
    #pragma unroll
    for (int kk = 0; kk < TK; kk += 8)
      Bs[bn][bk + kk] = cvt_bf16(W[(size_t)(k0 + bk + kk) * D + n0 + bn]);
    __syncthreads();
    v16bf a, b0, b1;
    #pragma unroll
    for (int i = 0; i < 16; ++i) {
      const int k = kb + ((i < 8) ? i : (i + 8));
      a[i]  = As[mloc][k];
      b0[i] = Bs[nloc][k];
      b1[i] = Bs[nloc + 16][k];
    }
    c0 = __builtin_amdgcn_wmma_f32_16x16x32_bf16(false, a, false, b0,
                                                 (short)0, c0, false, false);
    c1 = __builtin_amdgcn_wmma_f32_16x16x32_bf16(false, a, false, b1,
                                                 (short)0, c1, false, false);
    __syncthreads();
  }
  const int rbase = wave * 16 + ((lane >= 16) ? 8 : 0);
  #pragma unroll
  for (int r = 0; r < 8; ++r) {
    const int m  = Mbase + rbase + r;
    const int bb = m >> 8;            // m / S
    const int t  = m & (S - 1);       // m % S
    Ht[((size_t)bb * D + (n0 + nloc)) * S + t]      = cvt_bf16(c0[r]);
    Ht[((size_t)bb * D + (n0 + 16 + nloc)) * S + t] = cvt_bf16(c1[r]);
  }
}

// GEMM2: Xout[b,s,e] = relu( (adj[b] @ H[b])[s,e] / denom[b,s] + bias[e] )
// B-panel (bf16, K-contiguous rows of Ht) is DMA'd to LDS asynchronously.
__global__ __launch_bounds__(256) void gcn_adj_kernel(
    const float* __restrict__ adj, const __bf16* __restrict__ Ht,
    const float* __restrict__ denomv, const float* __restrict__ bias,
    float* __restrict__ Xout) {
  __shared__ __bf16 As[TM][LDA];
  __shared__ __bf16 Bs[TN][LDB];
  const int tid  = threadIdx.x;
  const int lane = tid & 31, wave = tid >> 5;
  const int b     = blockIdx.x >> 1;
  const int Mbase = (blockIdx.x & 1) * TM;
  const int n0    = blockIdx.y * TN;

  const int arow  = tid >> 1;
  const int acol0 = (tid & 1) * 16;
  const float* Arow = adj + ((size_t)b * S + (Mbase + arow)) * S;

  // async B mapping: 32 rows x 64B, 8 bytes (4 bf16) per lane
  const int an = tid >> 3;          // 0..31
  const int aq = (tid & 7) * 4;     // half-index 0,4,...,28
  const unsigned long long Hb =
      (unsigned long long)(Ht + ((size_t)b * D + n0) * S);  // uniform
  const unsigned bs_lds = lds_off32(&Bs[an][aq]);

  const int kb   = (lane < 16) ? 0 : 8;
  const int mloc = wave * 16 + (lane & 15);
  const int nloc = lane & 15;

  v8f c0 = {}, c1 = {};
  for (int k0 = 0; k0 < S; k0 += TK) {
    // B panel: global -> LDS DMA, 8B per lane (GVS: SGPR64 base + VGPR32 off)
    {
      const unsigned goff = (unsigned)(((an * S) + k0 + aq) * 2);
      asm volatile("global_load_async_to_lds_b64 %0, %1, %2"
                   :: "v"(bs_lds), "v"(goff), "s"(Hb) : "memory");
    }
    if (k0 + TK < S) __builtin_prefetch(Arow + k0 + TK + acol0, 0, 0);
    #pragma unroll
    for (int q = 0; q < 4; ++q) {
      float4 v = *(const float4*)(Arow + k0 + acol0 + 4 * q);
      As[arow][acol0 + 4 * q + 0] = cvt_bf16(v.x);
      As[arow][acol0 + 4 * q + 1] = cvt_bf16(v.y);
      As[arow][acol0 + 4 * q + 2] = cvt_bf16(v.z);
      As[arow][acol0 + 4 * q + 3] = cvt_bf16(v.w);
    }
    asm volatile("s_wait_asynccnt 0" ::: "memory");
    __syncthreads();
    v16bf a, b0, b1;
    #pragma unroll
    for (int i = 0; i < 16; ++i) {
      const int k = kb + ((i < 8) ? i : (i + 8));
      a[i]  = As[mloc][k];
      b0[i] = Bs[nloc][k];
      b1[i] = Bs[nloc + 16][k];
    }
    c0 = __builtin_amdgcn_wmma_f32_16x16x32_bf16(false, a, false, b0,
                                                 (short)0, c0, false, false);
    c1 = __builtin_amdgcn_wmma_f32_16x16x32_bf16(false, a, false, b1,
                                                 (short)0, c1, false, false);
    __syncthreads();
  }
  const int rbase = wave * 16 + ((lane >= 16) ? 8 : 0);
  const float bv0 = bias[n0 + nloc];
  const float bv1 = bias[n0 + 16 + nloc];
  #pragma unroll
  for (int r = 0; r < 8; ++r) {
    const int s = Mbase + rbase + r;
    const float rd = 1.0f / denomv[b * S + s];
    float v0 = fmaxf(c0[r] * rd + bv0, 0.0f);
    float v1 = fmaxf(c1[r] * rd + bv1, 0.0f);
    Xout[((size_t)b * S + s) * D + n0 + nloc]      = v0;
    Xout[((size_t)b * S + s) * D + n0 + 16 + nloc] = v1;
  }
}

// ---------------------------------------------------------------------------
// Kernel 5: xs[b,d] = sum_{s=start..end} x3[b,s,d]   (mask collapses to rows)
// ---------------------------------------------------------------------------
__global__ __launch_bounds__(256) void attn1_kernel(
    const float* __restrict__ X3, const int* __restrict__ stA,
    const int* __restrict__ enA, float* __restrict__ xs) {
  const int b = blockIdx.x, tid = threadIdx.x;
  const int st = stA[b], en = enA[b];
  for (int d = tid; d < D; d += 256) {
    float acc = 0.0f;
    for (int s = st; s <= en; ++s) acc += X3[((size_t)b * S + s) * D + d];
    xs[b * D + d] = acc;
  }
}

// ---------------------------------------------------------------------------
// Kernel 6: scores -> softmax -> pooled -> FC.  One block per batch.
// ---------------------------------------------------------------------------
__global__ __launch_bounds__(S) void attn2_kernel(
    const float* __restrict__ xs, const float* __restrict__ text_out,
    const float* __restrict__ fcw, const float* __restrict__ fcb,
    float* __restrict__ out) {
  const int b = blockIdx.x, t = threadIdx.x;
  __shared__ float red[S];
  __shared__ float alpha[S];
  __shared__ float pooled[D];
  const float* xv  = xs + (size_t)b * D;
  const float* row = text_out + ((size_t)b * S + t) * D;
  float sc = 0.0f;
  for (int d = 0; d < D; ++d) sc += xv[d] * row[d];
  red[t] = sc;
  __syncthreads();
  for (int off = S / 2; off; off >>= 1) {
    if (t < off) red[t] = fmaxf(red[t], red[t + off]);
    __syncthreads();
  }
  const float mx = red[0];
  __syncthreads();
  const float e = __expf(sc - mx);
  red[t] = e;
  __syncthreads();
  for (int off = S / 2; off; off >>= 1) {
    if (t < off) red[t] += red[t + off];
    __syncthreads();
  }
  alpha[t] = e / red[0];
  __syncthreads();
  for (int d = t; d < D; d += S) {
    float acc = 0.0f;
    for (int tt = 0; tt < S; ++tt)
      acc += alpha[tt] * text_out[((size_t)b * S + tt) * D + d];
    pooled[d] = acc;
  }
  __syncthreads();
  if (t < P) {
    float acc = fcb[t];
    for (int d = 0; d < D; ++d) acc += pooled[d] * fcw[t * D + d];
    out[b * P + t] = acc;
  }
}

// ---------------------------------------------------------------------------
extern "C" void kernel_launch(void* const* d_in, const int* in_sizes, int n_in,
                              void* d_out, int out_size, void* d_ws,
                              size_t ws_size, hipStream_t stream) {
  const float* text_out = (const float*)d_in[0];
  const float* adj      = (const float*)d_in[1];
  const int*   ti       = (const int*)d_in[2];
  const int*   ai       = (const int*)d_in[3];
  const int*   li       = (const int*)d_in[4];
  const float* W1 = (const float*)d_in[5];
  const float* b1 = (const float*)d_in[6];
  const float* W2 = (const float*)d_in[7];
  const float* b2 = (const float*)d_in[8];
  const float* W3 = (const float*)d_in[9];
  const float* b3 = (const float*)d_in[10];
  const float* fcw = (const float*)d_in[11];
  const float* fcb = (const float*)d_in[12];
  float* out = (float*)d_out;

  char* ws = (char*)d_ws;
  size_t off = 0;
  auto alloc = [&](size_t bytes) -> void* {
    void* p = ws + off;
    off += (bytes + 255) & ~(size_t)255;
    return p;
  };
  float*  pw     = (float*)alloc((size_t)NB * S * 4);
  float*  denomv = (float*)alloc((size_t)NB * S * 4);
  int*    stA    = (int*)alloc((size_t)NB * 4);
  int*    enA    = (int*)alloc((size_t)NB * 4);
  float*  xs     = (float*)alloc((size_t)NB * D * 4);
  __bf16* Ht     = (__bf16*)alloc((size_t)NB * D * S * 2);
  float*  xA     = (float*)alloc((size_t)NB * S * D * 4);
  float*  xB     = (float*)alloc((size_t)NB * S * D * 4);

  meta_kernel<<<NB, S, 0, stream>>>(ti, ai, li, pw, stA, enA);
  denom_kernel<<<(NB * S) / 8, 256, 0, stream>>>(adj, denomv);

  const dim3 g((NB * S) / TM, D / TN), blk(256);
  gcn_xw_kernel<<<g, blk, 0, stream>>>(text_out, W1, pw, Ht);
  gcn_adj_kernel<<<g, blk, 0, stream>>>(adj, Ht, denomv, b1, xA);
  gcn_xw_kernel<<<g, blk, 0, stream>>>(xA, W2, pw, Ht);
  gcn_adj_kernel<<<g, blk, 0, stream>>>(adj, Ht, denomv, b2, xB);
  gcn_xw_kernel<<<g, blk, 0, stream>>>(xB, W3, pw, Ht);
  gcn_adj_kernel<<<g, blk, 0, stream>>>(adj, Ht, denomv, b3, xA);

  attn1_kernel<<<NB, 256, 0, stream>>>(xA, stA, enA, xs);
  attn2_kernel<<<NB, S, 0, stream>>>(xs, text_out, fcw, fcb, out);
}
